// SS_Conv_SSM_44744969290135
// MI455X (gfx1250) — matrix operand order repaired and checked by reading, not compiled, not run on hardware
//
#include <hip/hip_runtime.h>
#include <hip/hip_bf16.h>

typedef __attribute__((ext_vector_type(16))) _Float16 v16h;
typedef __attribute__((ext_vector_type(8)))  _Float16 v8h;
typedef __attribute__((ext_vector_type(8)))  float    v8f;

constexpr int Bc = 8, Hc = 56, Wc = 56, Cc = 192, Lc = 56 * 56;
constexpr int DH = 96, DI = 192, NS = 16, DR = 6;
constexpr int PSTR = 40;  // padded proj row stride (floats): dts@0..5, B@8..23, C@24..39

__device__ __forceinline__ v8f wmma_f16(v16h a, v16h b, v8f c) {
  // 8 args: (neg_a, A, neg_b, B, c_mod, C, reuse_a, reuse_b)
  return __builtin_amdgcn_wmma_f32_16x16x32_f16(false, a, false, b, (short)0, c,
                                                false, false);
}

// 16-bit A-matrix 16x32: lane<16 holds K={0..7,16..23}, lane>=16 holds K={8..15,24..31}.
// A lane's 16 values are two contiguous 8xf16 runs -> two b128 loads.
__device__ __forceinline__ v16h load_a_frag(const _Float16* row, int kb, int lane) {
  const int base = kb + ((lane < 16) ? 0 : 8);
  const v8h lo = *(const v8h*)(row + base);
  const v8h hi = *(const v8h*)(row + base + 16);
  v16h a;
#pragma unroll
  for (int e = 0; e < 8; ++e) { a[e] = lo[e]; a[e + 8] = hi[e]; }
  return a;
}

// 16-bit B-matrix 32x16: lanes 0-15 hold K={0..7,16..23} (N=lane), lanes 16-31 hold K={8..15,24..31}.
// With K-contiguous f16 weights this is two contiguous b128 loads per lane.
__device__ __forceinline__ v16h load_b_frag(const _Float16* col, int kb, int lane) {
  const int base = kb + ((lane < 16) ? 0 : 8);
  const v8h lo = *(const v8h*)(col + base);
  const v8h hi = *(const v8h*)(col + base + 16);
  v16h b;
#pragma unroll
  for (int e = 0; e < 8; ++e) { b[e] = lo[e]; b[e + 8] = hi[e]; }
  return b;
}

// ---------------------------------------------------------------------------
// Weight prep (once per launch): f32 -> f16, conv weights transposed to
// [tap][co][ci] so the WMMA K-dim is contiguous; x_proj padded to N=48.
// ---------------------------------------------------------------------------
__global__ void cvt_f16_kernel(const float* __restrict__ src, _Float16* __restrict__ dst,
                               int n) {
  const int i = blockIdx.x * 256 + threadIdx.x;
  if (i < n) dst[i] = (_Float16)src[i];
}

__global__ void cvt_conv_w_kernel(const float* __restrict__ src, _Float16* __restrict__ dst,
                                  int ntap) {  // src[tap][ci][co] -> dst[tap][co][ci]
  const int i = blockIdx.x * 256 + threadIdx.x;
  if (i >= ntap * DH * DH) return;
  const int co = i % DH, ci = (i / DH) % DH, tap = i / (DH * DH);
  dst[(tap * DH + co) * DH + ci] = (_Float16)src[(tap * DH + ci) * DH + co];
}

__global__ void cvt_xproj_w_kernel(const float* __restrict__ src,  // (4,38,192)
                                   _Float16* __restrict__ dst) {   // (4,48,192), zero pad
  const int i = blockIdx.x * 256 + threadIdx.x;
  if (i >= 4 * 48 * DI) return;
  const int c = i % DI, n = (i / DI) % 48, k = i / (DI * 48);
  dst[i] = (n < 38) ? (_Float16)src[((size_t)(k * 38 + n)) * DI + c] : (_Float16)0.f;
}

// ---------------------------------------------------------------------------
// Conv path: BN(+opt ReLU) folded to scale/shift, halo staged to LDS as f16,
// implicit-GEMM conv via WMMA. Block = 256 thr (8 waves) = one (b,h) row,
// 24 tiles (4 m x 6 co), 3 tiles/wave, fully unrolled 27 k-steps for 3x3.
// ---------------------------------------------------------------------------
template <int THREE>
__global__ void __launch_bounds__(256)
conv_bn_wmma(const float* __restrict__ in, int in_stride,
             const float* __restrict__ g, const float* __restrict__ bb,
             const float* __restrict__ mm, const float* __restrict__ vv,
             int relu_in, const _Float16* __restrict__ wt,  // [tap][co][ci] f16
             const float* __restrict__ bias, int relu_out,
             float* __restrict__ out) {
  constexpr int NR = THREE ? 3 : 1;
  __shared__ __align__(16) _Float16 sIn[NR][68][DH];
  __shared__ float sScale[DH], sShift[DH];
  const int t = threadIdx.x;
  const int b = blockIdx.x / Hc, h = blockIdx.x % Hc;

  if (t < DH) {
    const float s = g[t] * rsqrtf(vv[t] + 1e-5f);
    sScale[t] = s;
    sShift[t] = bb[t] - mm[t] * s;
  }
  __syncthreads();

  for (int i = t; i < NR * 68 * DH; i += 256) {
    const int ci = i % DH;
    const int c  = (i / DH) % 68;
    const int r  = i / (DH * 68);
    const int hh = h + (THREE ? r - 1 : 0);
    const int ww = c - (THREE ? 1 : 0);
    float val = 0.f;  // SAME padding is zero AFTER bn
    if (hh >= 0 && hh < Hc && ww >= 0 && ww < Wc) {
      val = in[(size_t)((b * Hc + hh) * Wc + ww) * in_stride + ci] * sScale[ci] + sShift[ci];
      if (relu_in) val = fmaxf(val, 0.f);
    }
    sIn[r][c][ci] = (_Float16)val;
  }
  __syncthreads();

  const int lane = t & 31, wv = t >> 5;
  const int Mrow = lane & 15;
  for (int i = 0; i < 3; ++i) {
    const int ti = wv + 8 * i;       // 24 tiles: 4 m x 6 n
    const int mt = ti & 3, nt = ti >> 2;
    const int wposA = mt * 16 + Mrow;
    const int co = nt * 16 + Mrow;
    v8f acc = {};
#pragma unroll
    for (int dy = 0; dy < (THREE ? 3 : 1); ++dy) {
#pragma unroll
      for (int dx = 0; dx < (THREE ? 3 : 1); ++dx) {
        const int c = THREE ? (wposA + dx) : wposA;
        const _Float16* wcol =
            wt + ((size_t)(THREE ? (dy * 3 + dx) : 0) * DH + co) * DH;
#pragma unroll
        for (int kb = 0; kb < DH; kb += 32) {
          const v16h A = load_a_frag(&sIn[dy][c][0], kb, lane);
          const v16h Bf = load_b_frag(wcol, kb, lane);
          acc = wmma_f16(A, Bf, acc);
        }
      }
    }
    const float bi = bias[co];
#pragma unroll
    for (int r = 0; r < 8; ++r) {
      const int M = r + ((lane < 16) ? 0 : 8);
      const int wp = mt * 16 + M;
      if (wp < Wc) {
        float val = acc[r] + bi;
        if (relu_out) val = fmaxf(val, 0.f);
        out[(size_t)((b * Hc + h) * Wc + wp) * DH + co] = val;
      }
    }
  }
}

// ---------------------------------------------------------------------------
// LayerNorm(xr) + in_proj GEMM (M=64 positions/block, N=384, K=96). WMMA.
// ---------------------------------------------------------------------------
__global__ void __launch_bounds__(256)
ln_inproj_wmma(const float* __restrict__ x,
               const float* __restrict__ lng, const float* __restrict__ lnb,
               const _Float16* __restrict__ Wp,  // (384,96) f16
               float* __restrict__ xi, float* __restrict__ zb) {
  __shared__ __align__(16) _Float16 sA[64][DH];
  const int t = threadIdx.x;
  const size_t pos0 = (size_t)blockIdx.x * 64;
  if (t < 64) {
    const float* xr = x + (pos0 + t) * Cc + DH;
    float mu = 0.f;
    for (int c = 0; c < DH; ++c) mu += xr[c];
    mu *= (1.f / DH);
    float var = 0.f;
    for (int c = 0; c < DH; ++c) { float d = xr[c] - mu; var += d * d; }
    var *= (1.f / DH);
    const float is = rsqrtf(var + 1e-5f);
    for (int c = 0; c < DH; ++c)
      sA[t][c] = (_Float16)((xr[c] - mu) * is * lng[c] + lnb[c]);
  }
  __syncthreads();
  const int lane = t & 31, wv = t >> 5;
  for (int i = 0; i < 12; ++i) {
    const int ti = wv + 8 * i;           // 96 tiles: 4 m-tiles x 24 n-tiles
    const int mt = ti & 3, nt = ti >> 2;
    const int n = nt * 16 + (lane & 15); // output feature 0..383
    v8f acc = {};
#pragma unroll
    for (int kb = 0; kb < DH; kb += 32) {
      const v16h A = load_a_frag(&sA[mt * 16 + (lane & 15)][0], kb, lane);
      const v16h Bf = load_b_frag(Wp + (size_t)n * DH, kb, lane);
      acc = wmma_f16(A, Bf, acc);
    }
#pragma unroll
    for (int r = 0; r < 8; ++r) {
      const int M = r + ((lane < 16) ? 0 : 8);
      const size_t pos = pos0 + mt * 16 + M;
      if (n < DI) xi[pos * DI + n] = acc[r];
      else        zb[pos * DI + (n - DI)] = acc[r];
    }
  }
}

// ---------------------------------------------------------------------------
// Depthwise 3x3 conv + SiLU (plain VALU)
// ---------------------------------------------------------------------------
__global__ void dwconv_silu(const float* __restrict__ xi,
                            const float* __restrict__ cw, const float* __restrict__ cb,
                            float* __restrict__ xc) {
  const size_t idx = (size_t)blockIdx.x * blockDim.x + threadIdx.x;
  if (idx >= (size_t)Bc * Lc * DI) return;
  const int d = idx % DI;
  const size_t p = idx / DI;
  const int w = p % Wc, h = (p / Wc) % Hc, b = p / ((size_t)Hc * Wc);
  float s = cb[d];
  for (int dy = 0; dy < 3; ++dy) {
    const int hh = h + dy - 1;
    if (hh < 0 || hh >= Hc) continue;
    for (int dx = 0; dx < 3; ++dx) {
      const int ww = w + dx - 1;
      if (ww < 0 || ww >= Wc) continue;
      s += xi[((size_t)(b * Hc + hh) * Wc + ww) * DI + d] * cw[(dy * 3 + dx) * DI + d];
    }
  }
  xc[idx] = s / (1.f + expf(-s));  // silu
}

// ---------------------------------------------------------------------------
// x_proj: per-position, per-direction 38x192 projection. WMMA, N padded to 48
// with zero-padded f16 weights. proj rows padded to PSTR=40 floats:
// dts@[0..5], B@[8..23], C@[24..39].
// ---------------------------------------------------------------------------
__global__ void __launch_bounds__(256)
xproj_wmma(const float* __restrict__ xc,
           const _Float16* __restrict__ Wx,  // (4,48,192) f16, zero-padded
           float* __restrict__ proj) {       // (B,4,L,PSTR)
  __shared__ __align__(16) _Float16 sA[64][DI];
  const int t = threadIdx.x;
  const size_t pos0 = (size_t)blockIdx.x * 64;
  for (int i = t; i < 64 * DI; i += 256) {
    const int row = i / DI, c = i % DI;
    sA[row][c] = (_Float16)xc[(pos0 + row) * DI + c];
  }
  __syncthreads();
  const int lane = t & 31, wv = t >> 5;
  for (int i = 0; i < 6; ++i) {
    const int ti = wv + 8 * i;           // 48 tiles: 4 m x (4 dirs x 3 n)
    const int mt = ti & 3, rest = ti >> 2;
    const int k = rest / 3, nt = rest % 3;
    const int n = nt * 16 + (lane & 15);
    v8f acc = {};
#pragma unroll
    for (int kb = 0; kb < DI; kb += 32) {
      const v16h A = load_a_frag(&sA[mt * 16 + (lane & 15)][0], kb, lane);
      const v16h Bf = load_b_frag(Wx + (size_t)(k * 48 + n) * DI, kb, lane);
      acc = wmma_f16(A, Bf, acc);
    }
    if (n < 38) {
      const int slot = (n < 6) ? n : (n + 2);  // align B/C regions to 16B
#pragma unroll
      for (int r = 0; r < 8; ++r) {
        const int M = r + ((lane < 16) ? 0 : 8);
        const size_t pos = pos0 + mt * 16 + M;
        const size_t bidx = pos / Lc, psp = pos % Lc;
        proj[(((bidx * 4 + k) * Lc) + psp) * PSTR + slot] = acc[r];
      }
    }
  }
}

__global__ void zero_kernel(float* __restrict__ p, size_t n) {
  const size_t i = (size_t)blockIdx.x * blockDim.x + threadIdx.x;
  if (i < n) p[i] = 0.f;
}

// ---------------------------------------------------------------------------
// Selective scan: one block per (batch, direction); thread = channel d.
// 16 states in registers; direction flips/transposes are index math; output
// accumulated with global_atomic_add_f32 into shared yacc buffer.
// ---------------------------------------------------------------------------
__global__ void scan_kernel(const float* __restrict__ proj, const float* __restrict__ xc,
                            const float* __restrict__ dtw,  // (4,192,6)
                            const float* __restrict__ dtb,  // (4,192)
                            const float* __restrict__ Alog, // (4,192,16)
                            const float* __restrict__ Dsp,  // (4,192)
                            float* __restrict__ yacc) {
  const int bk = blockIdx.x;
  const int b = bk >> 2, k = bk & 3;
  const int d = threadIdx.x;  // 0..191
  float dw[DR];
#pragma unroll
  for (int r = 0; r < DR; ++r) dw[r] = dtw[((size_t)(k * DI) + d) * DR + r];
  const float db = dtb[k * DI + d];
  float An[NS];
#pragma unroll
  for (int n = 0; n < NS; ++n) An[n] = -expf(Alog[((size_t)(k * DI) + d) * NS + n]);
  const float Dk = Dsp[k * DI + d];
  float hs[NS];
#pragma unroll
  for (int n = 0; n < NS; ++n) hs[n] = 0.f;

  const float* pbase = proj + (size_t)(b * 4 + k) * Lc * PSTR;
  const float* ubase = xc + (size_t)b * Lc * DI;
  float* ybase = yacc + (size_t)b * Lc * DI;

  for (int l = 0; l < Lc; ++l) {
    int p;
    if (k == 0)      p = l;
    else if (k == 1) { const int w_ = l / Hc, h_ = l % Hc; p = h_ * Wc + w_; }
    else if (k == 2) p = Lc - 1 - l;
    else             { const int lp = Lc - 1 - l; const int w_ = lp / Hc, h_ = lp % Hc; p = h_ * Wc + w_; }
    const float* pr =
        (const float*)__builtin_assume_aligned(pbase + (size_t)p * PSTR, 16);
    __builtin_prefetch((const void*)(pr + PSTR), 0, 1);  // gfx1250 global_prefetch_b8
    float dp = db;
#pragma unroll
    for (int r = 0; r < DR; ++r) dp += dw[r] * pr[r];
    const float delta = (dp > 20.f) ? dp : log1pf(expf(dp));  // softplus
    const float u = ubase[(size_t)p * DI + d];
    const float du = delta * u;
    float y = Dk * u;
#pragma unroll
    for (int n = 0; n < NS; ++n) {
      hs[n] = expf(delta * An[n]) * hs[n] + du * pr[8 + n];
      y += hs[n] * pr[24 + n];
    }
    atomicAdd(&ybase[(size_t)p * DI + d], y);
  }
}

// ---------------------------------------------------------------------------
// out_norm(yv) * silu(z), out_proj GEMM (N=96, K=192) via WMMA, interleave odd
// output channels + residual.
// ---------------------------------------------------------------------------
__global__ void __launch_bounds__(256)
merge_outproj_wmma(const float* __restrict__ yacc, const float* __restrict__ zb,
                   const float* __restrict__ ong, const float* __restrict__ onb,
                   const _Float16* __restrict__ Wo,  // (96,192) f16
                   const float* __restrict__ x, float* __restrict__ out) {
  __shared__ __align__(16) _Float16 sA[64][DI];
  const int t = threadIdx.x;
  const size_t pos0 = (size_t)blockIdx.x * 64;
  if (t < 64) {
    const float* yv = yacc + (pos0 + t) * DI;
    float mu = 0.f;
    for (int e = 0; e < DI; ++e) mu += yv[e];
    mu *= (1.f / DI);
    float var = 0.f;
    for (int e = 0; e < DI; ++e) { float d = yv[e] - mu; var += d * d; }
    var *= (1.f / DI);
    const float is = rsqrtf(var + 1e-5f);
    for (int e = 0; e < DI; ++e) {
      const float g = (yv[e] - mu) * is * ong[e] + onb[e];
      const float zz = zb[(pos0 + t) * DI + e];
      sA[t][e] = (_Float16)(g * (zz / (1.f + expf(-zz))));
    }
  }
  __syncthreads();
  const int lane = t & 31, wv = t >> 5;
  for (int i = 0; i < 3; ++i) {
    const int ti = wv + 8 * i;           // 24 tiles: 4 m x 6 n
    const int mt = ti & 3, nt = ti >> 2;
    const int c = nt * 16 + (lane & 15); // out channel 0..95
    v8f acc = {};
#pragma unroll
    for (int kb = 0; kb < DI; kb += 32) {
      const v16h A = load_a_frag(&sA[mt * 16 + (lane & 15)][0], kb, lane);
      const v16h Bf = load_b_frag(Wo + (size_t)c * DI, kb, lane);
      acc = wmma_f16(A, Bf, acc);
    }
#pragma unroll
    for (int r = 0; r < 8; ++r) {
      const int M = r + ((lane < 16) ? 0 : 8);
      const size_t oi = (pos0 + mt * 16 + M) * Cc + 2 * c + 1;
      out[oi] = acc[r] + x[oi];
    }
  }
}

// conv-path half into even output channels + residual
__global__ void xl_merge(const float* __restrict__ xl_out, const float* __restrict__ x,
                         float* __restrict__ out) {
  const size_t idx = (size_t)blockIdx.x * blockDim.x + threadIdx.x;
  if (idx >= (size_t)Bc * Lc * DH) return;
  const size_t pos = idx / DH;
  const int i = idx % DH;
  const size_t oi = pos * Cc + 2 * i;
  out[oi] = xl_out[pos * DH + i] + x[oi];
}

extern "C" void kernel_launch(void* const* d_in, const int* in_sizes, int n_in,
                              void* d_out, int out_size, void* d_ws, size_t ws_size,
                              hipStream_t stream) {
  (void)in_sizes; (void)n_in; (void)out_size; (void)ws_size;
  const float* x    = (const float*)d_in[0];
  const float* bn1g = (const float*)d_in[1],  *bn1b = (const float*)d_in[2];
  const float* bn1m = (const float*)d_in[3],  *bn1v = (const float*)d_in[4];
  const float* bn2g = (const float*)d_in[5],  *bn2b = (const float*)d_in[6];
  const float* bn2m = (const float*)d_in[7],  *bn2v = (const float*)d_in[8];
  const float* bn3g = (const float*)d_in[9],  *bn3b = (const float*)d_in[10];
  const float* bn3m = (const float*)d_in[11], *bn3v = (const float*)d_in[12];
  const float* c1w = (const float*)d_in[13], *c1b = (const float*)d_in[14];
  const float* c2w = (const float*)d_in[15], *c2b = (const float*)d_in[16];
  const float* c3w = (const float*)d_in[17], *c3b = (const float*)d_in[18];
  const float* lng = (const float*)d_in[19], *lnb = (const float*)d_in[20];
  const float* inpw = (const float*)d_in[21];
  const float* dwc_w = (const float*)d_in[22], *dwc_b = (const float*)d_in[23];
  const float* xprw = (const float*)d_in[24];
  const float* dtpw = (const float*)d_in[25], *dtpb = (const float*)d_in[26];
  const float* Alg = (const float*)d_in[27], *Dsp = (const float*)d_in[28];
  const float* ong = (const float*)d_in[29], *onb = (const float*)d_in[30];
  const float* outw = (const float*)d_in[31];
  float* out = (float*)d_out;

  float* ws = (float*)d_ws;
  const size_t S1 = (size_t)Bc * Lc * DH;   // 2,408,448
  const size_t S2 = 2 * S1;                 // B*L*192
  const size_t SPROJ = (size_t)Bc * 4 * Lc * PSTR;
  float* y1     = ws;
  float* y2     = ws + S1;
  float* xl_out = ws + 2 * S1;
  float* xi     = ws + 3 * S1;
  float* zb     = ws + 3 * S1 + S2;
  float* xconv  = ws + 3 * S1 + 2 * S2;
  float* proj   = ws + 3 * S1 + 3 * S2;
  float* yacc   = xi;                       // xi dead after dwconv; reuse

  // f16 weight region after proj
  _Float16* h16 = (_Float16*)(proj + SPROJ);
  _Float16* w1t    = h16;                     // 9*96*96
  _Float16* w2t    = w1t + 9 * DH * DH;       // 9*96*96
  _Float16* w3t    = w2t + 9 * DH * DH;       // 1*96*96
  _Float16* inpw16 = w3t + DH * DH;           // 384*96
  _Float16* xprw16 = inpw16 + 2 * DI * DH;    // 4*48*192 (zero-padded)
  _Float16* outw16 = xprw16 + 4 * 48 * DI;    // 96*192

  // weight prep (cheap, deterministic per call)
  cvt_conv_w_kernel<<<(9 * DH * DH + 255) / 256, 256, 0, stream>>>(c1w, w1t, 9);
  cvt_conv_w_kernel<<<(9 * DH * DH + 255) / 256, 256, 0, stream>>>(c2w, w2t, 9);
  cvt_conv_w_kernel<<<(DH * DH + 255) / 256, 256, 0, stream>>>(c3w, w3t, 1);
  cvt_f16_kernel<<<(2 * DI * DH + 255) / 256, 256, 0, stream>>>(inpw, inpw16, 2 * DI * DH);
  cvt_xproj_w_kernel<<<(4 * 48 * DI + 255) / 256, 256, 0, stream>>>(xprw, xprw16);
  cvt_f16_kernel<<<(DH * DI + 255) / 256, 256, 0, stream>>>(outw, outw16, DH * DI);

  // conv half-path
  conv_bn_wmma<1><<<Bc * Hc, 256, 0, stream>>>(x, Cc, bn1g, bn1b, bn1m, bn1v, 0, w1t, c1b, 0, y1);
  conv_bn_wmma<1><<<Bc * Hc, 256, 0, stream>>>(y1, DH, bn2g, bn2b, bn2m, bn2v, 1, w2t, c2b, 0, y2);
  conv_bn_wmma<0><<<Bc * Hc, 256, 0, stream>>>(y2, DH, bn3g, bn3b, bn3m, bn3v, 1, w3t, c3b, 1, xl_out);
  // ssm half-path
  const int nblk64 = (Bc * Lc) / 64;  // 392
  ln_inproj_wmma<<<nblk64, 256, 0, stream>>>(x, lng, lnb, inpw16, xi, zb);
  dwconv_silu<<<(unsigned)(((size_t)Bc * Lc * DI + 255) / 256), 256, 0, stream>>>(xi, dwc_w, dwc_b, xconv);
  xproj_wmma<<<nblk64, 256, 0, stream>>>(xconv, xprw16, proj);
  zero_kernel<<<(unsigned)((S2 + 255) / 256), 256, 0, stream>>>(yacc, S2);
  scan_kernel<<<Bc * 4, DI, 0, stream>>>(proj, xconv, dtpw, dtpb, Alg, Dsp, yacc);
  merge_outproj_wmma<<<nblk64, 256, 0, stream>>>(yacc, zb, ong, onb, outw16, x, out);
  xl_merge<<<(unsigned)(((size_t)Bc * Lc * DH + 255) / 256), 256, 0, stream>>>(xl_out, x, out);
}
